// VQCCell_42082089566657
// MI455X (gfx1250) — matrix operand order repaired and checked by reading, not compiled, not run on hardware
//
#include <hip/hip_runtime.h>
#include <cstdint>

typedef __attribute__((ext_vector_type(16))) _Float16 v16h;
typedef __attribute__((ext_vector_type(8)))  float    v8f;

#define NQ   6
#define DIM  64      // 2^NQ
#define KIN  256
#define NOUT 128

// Build a 16-element f16 fragment from four float4s, scaled (scale=0 zero-pads).
__device__ __forceinline__ v16h cvt16(const float4 a, const float4 b,
                                      const float4 c, const float4 d,
                                      const float scale) {
  float t[16] = {a.x, a.y, a.z, a.w, b.x, b.y, b.z, b.w,
                 c.x, c.y, c.z, c.w, d.x, d.y, d.z, d.w};
  v16h f;
#pragma unroll
  for (int i = 0; i < 16; ++i) f[i] = (_Float16)(t[i] * scale);
  return f;
}

// A fragment (16-bit A 16x32 layout): this lane's row, K = k0..k0+7 and k0+16..k0+23.
__device__ __forceinline__ v16h load_a_frag(const float* __restrict__ rowp) {
  const float4* p = reinterpret_cast<const float4*>(rowp);
  float4 x0 = p[0], x1 = p[1];   // k0 .. k0+7
  float4 y0 = p[4], y1 = p[5];   // k0+16 .. k0+23
  return cvt16(x0, x1, y0, y1, 1.0f);
}

__global__ __launch_bounds__(256)
void vqc_fused(const float* __restrict__ f_bt,
               const float* __restrict__ Wr,
               const float* __restrict__ br,
               const float* __restrict__ qweights,
               const float* __restrict__ Wp,
               const float* __restrict__ bp,
               float* __restrict__ out,
               int nrows) {
  __shared__ float xfer[256][8];   // per-block z-preact transpose buffer

  const int lane = threadIdx.x & 31;
  const int wv   = threadIdx.x >> 5;            // wave id in block (0..7)
  const int rowbase = blockIdx.x * 256 + wv * 32;
  if (rowbase >= nrows) return;

  const int   n      = lane & 15;               // B/C column (N)
  const bool  bvalid = (n < NQ);
  const int   nc     = bvalid ? n : 0;
  const float bmask  = bvalid ? 1.0f : 0.0f;
  const int   khalfB = (lane < 16) ? 0 : 8;     // B: lanes 0-15 K=0..7,16..23; 16-31 K=8..15,24..31
  const int   khalfA = (lane < 16) ? 0 : 8;     // A: lanes 0-15 K base 0, 16-31 base 8

  // ---- B fragments: Wr^T zero-padded to 32x16 per 32-K chunk, f16 ----
  v16h bfrag[8];
  {
    const float* wrow = Wr + nc * KIN;
#pragma unroll
    for (int kb = 0; kb < 8; ++kb) {
      const float4* p = reinterpret_cast<const float4*>(wrow + kb * 32 + khalfB);
      bfrag[kb] = cvt16(p[0], p[1], p[4], p[5], bmask);
    }
  }

  // ---- Stage 1 GEMM: two 16x16 C tiles per wave, K=256 (8 WMMAs each) ----
  v8f acc0, acc1;
#pragma unroll
  for (int j = 0; j < 8; ++j) { acc0[j] = 0.0f; acc1[j] = 0.0f; }

  int r0 = rowbase + (lane & 15);
  int r1 = r0 + 16;
  r0 = (r0 < nrows) ? r0 : (nrows - 1);
  r1 = (r1 < nrows) ? r1 : (nrows - 1);
  const float* rp0 = f_bt + (size_t)r0 * KIN;
  const float* rp1 = f_bt + (size_t)r1 * KIN;

#pragma unroll
  for (int kb = 0; kb < 8; ++kb) {
    if (kb < 7) {
      __builtin_prefetch(rp0 + (kb + 1) * 32, 0, 1);
      __builtin_prefetch(rp1 + (kb + 1) * 32, 0, 1);
    }
    v16h a0 = load_a_frag(rp0 + kb * 32 + khalfA);
    v16h a1 = load_a_frag(rp1 + kb * 32 + khalfA);
    acc0 = __builtin_amdgcn_wmma_f32_16x16x32_f16(false, a0, false, bfrag[kb],
                                                  (short)0, acc0, false, false);
    acc1 = __builtin_amdgcn_wmma_f32_16x16x32_f16(false, a1, false, bfrag[kb],
                                                  (short)0, acc1, false, false);
  }

  // ---- Transpose C -> per-lane row ownership via LDS ----
  // C layout: VGPR j holds M=j (lanes 0-15) / M=8+j (lanes 16-31), N = lane%16.
  const int mbase = (lane < 16) ? 0 : 8;
  if (bvalid) {
#pragma unroll
    for (int j = 0; j < 8; ++j) {
      xfer[wv * 32 + mbase + j][n]      = acc0[j];
      xfer[wv * 32 + 16 + mbase + j][n] = acc1[j];
    }
  }
  __syncthreads();

  float z[NQ];
#pragma unroll
  for (int w = 0; w < NQ; ++w)
    z[w] = tanhf(xfer[wv * 32 + lane][w] + br[w]);

  // ---- Stage 2: VQC state-vector simulation, 64 complex amps in VGPRs ----
  float sr[DIM], si[DIM];
#pragma unroll
  for (int i = 0; i < DIM; ++i) { sr[i] = 0.0f; si[i] = 0.0f; }
  sr[0] = 1.0f;

  // Encoding rotations: real gate [[c,-s],[s,c]] on each qubit
#pragma unroll
  for (int q = 0; q < NQ; ++q) {
    const float h  = 0.5f * z[q];
    const float cq = __cosf(h), sq = __sinf(h);
    const int   m  = 1 << (5 - q);
#pragma unroll
    for (int i = 0; i < DIM; ++i) {
      if (!(i & m)) {
        const int i1 = i | m;
        const float ar = sr[i],  ai = si[i];
        const float cr = sr[i1], ci = si[i1];
        sr[i]  = cq * ar - sq * cr;  si[i]  = cq * ai - sq * ci;
        sr[i1] = sq * ar + cq * cr;  si[i1] = sq * ai + cq * ci;
      }
    }
  }

  // Two layers: Rot(phi,theta,omega) on each qubit, then CNOT ring
#pragma unroll
  for (int l = 0; l < 2; ++l) {
#pragma unroll
    for (int q = 0; q < NQ; ++q) {
      const float phi = qweights[(l * NQ + q) * 3 + 0];
      const float th  = qweights[(l * NQ + q) * 3 + 1];
      const float om  = qweights[(l * NQ + q) * 3 + 2];
      const float ct  = __cosf(0.5f * th), st = __sinf(0.5f * th);
      const float aa  = 0.5f * (phi + om), bb = 0.5f * (phi - om);
      const float epr = __cosf(aa), epi = -__sinf(aa);
      const float emr = __cosf(bb), emi = -__sinf(bb);
      // m00=ep*ct  m01=-conj(em)*st  m10=em*st  m11=conj(ep)*ct
      const float m00r =  epr * ct, m00i =  epi * ct;
      const float m01r = -emr * st, m01i =  emi * st;
      const float m10r =  emr * st, m10i =  emi * st;
      const float m11r =  epr * ct, m11i = -epi * ct;
      const int m = 1 << (5 - q);
#pragma unroll
      for (int i = 0; i < DIM; ++i) {
        if (!(i & m)) {
          const int i1 = i | m;
          const float ar = sr[i],  ai = si[i];
          const float cr = sr[i1], ci = si[i1];
          sr[i]  = m00r * ar - m00i * ai + m01r * cr - m01i * ci;
          si[i]  = m00r * ai + m00i * ar + m01r * ci + m01i * cr;
          sr[i1] = m10r * ar - m10i * ai + m11r * cr - m11i * ci;
          si[i1] = m10r * ai + m10i * ar + m11r * ci + m11i * cr;
        }
      }
    }
    const int r = (l % (NQ - 1)) + 1;
#pragma unroll
    for (int q = 0; q < NQ; ++q) {
      const int mc = 1 << (5 - q);
      const int mt = 1 << (5 - ((q + r) % NQ));
#pragma unroll
      for (int i = 0; i < DIM; ++i) {
        if ((i & mc) && !(i & mt)) {
          const int i1 = i | mt;
          float t;
          t = sr[i]; sr[i] = sr[i1]; sr[i1] = t;
          t = si[i]; si[i] = si[i1]; si[i1] = t;
        }
      }
    }
  }

  // Pauli-Z expectation values
  float qv[NQ];
#pragma unroll
  for (int w = 0; w < NQ; ++w) qv[w] = 0.0f;
#pragma unroll
  for (int i = 0; i < DIM; ++i) {
    const float p = sr[i] * sr[i] + si[i] * si[i];
#pragma unroll
    for (int w = 0; w < NQ; ++w)
      qv[w] += ((i >> (5 - w)) & 1) ? -p : p;
  }

  // ---- Stage 3: out = q @ Wp^T + bp, coalesced via wave32 shuffle broadcast ----
  const int col0 = lane * 4;            // this lane owns output cols 4l..4l+3
  float wp[4][NQ];
#pragma unroll
  for (int r4 = 0; r4 < 4; ++r4)
#pragma unroll
    for (int w = 0; w < NQ; ++w)
      wp[r4][w] = Wp[(col0 + r4) * NQ + w];
  const float4 bpv = reinterpret_cast<const float4*>(bp)[lane];

#pragma unroll 4
  for (int rr = 0; rr < 32; ++rr) {
    if (rowbase + rr < nrows) {
      const float q0 = __shfl(qv[0], rr, 32);
      const float q1 = __shfl(qv[1], rr, 32);
      const float q2 = __shfl(qv[2], rr, 32);
      const float q3 = __shfl(qv[3], rr, 32);
      const float q4 = __shfl(qv[4], rr, 32);
      const float q5 = __shfl(qv[5], rr, 32);
      float4 o = bpv;
      o.x += q0*wp[0][0] + q1*wp[0][1] + q2*wp[0][2] + q3*wp[0][3] + q4*wp[0][4] + q5*wp[0][5];
      o.y += q0*wp[1][0] + q1*wp[1][1] + q2*wp[1][2] + q3*wp[1][3] + q4*wp[1][4] + q5*wp[1][5];
      o.z += q0*wp[2][0] + q1*wp[2][1] + q2*wp[2][2] + q3*wp[2][3] + q4*wp[2][4] + q5*wp[2][5];
      o.w += q0*wp[3][0] + q1*wp[3][1] + q2*wp[3][2] + q3*wp[3][3] + q4*wp[3][4] + q5*wp[3][5];
      reinterpret_cast<float4*>(out + (size_t)(rowbase + rr) * NOUT)[lane] = o;
    }
  }
}

extern "C" void kernel_launch(void* const* d_in, const int* in_sizes, int n_in,
                              void* d_out, int out_size, void* d_ws, size_t ws_size,
                              hipStream_t stream) {
  const float* f_bt = (const float*)d_in[0];
  const float* Wr   = (const float*)d_in[1];
  const float* br   = (const float*)d_in[2];
  const float* qw   = (const float*)d_in[3];
  const float* Wp   = (const float*)d_in[4];
  const float* bp   = (const float*)d_in[5];
  float* out = (float*)d_out;

  const int nrows  = in_sizes[0] / KIN;          // 262144
  const int blocks = (nrows + 255) / 256;        // 256 rows per block (8 waves x 32 rows)
  vqc_fused<<<blocks, 256, 0, stream>>>(f_bt, Wr, br, qw, Wp, bp, out, nrows);
}
